// BaseModel_13752485281942
// MI455X (gfx1250) — matrix operand order repaired and checked by reading, not compiled, not run on hardware
//
#include <hip/hip_runtime.h>
#include <hip/hip_bf16.h>

typedef __attribute__((ext_vector_type(16))) __bf16 v16bf;
typedef __attribute__((ext_vector_type(8)))  float  v8f;

// ---------------------------------------------------------------------------
// Small utility kernels
// ---------------------------------------------------------------------------

__global__ void deg_init(float* __restrict__ deg, int n) {
    int i = blockIdx.x * blockDim.x + threadIdx.x;
    if (i < n) deg[i] = 1.0f;                      // self-loop
}

__global__ void deg_acc(const int* __restrict__ dst, float* __restrict__ deg, int E) {
    int e = blockIdx.x * blockDim.x + threadIdx.x;
    if (e < E) atomicAdd(&deg[dst[e]], 1.0f);
}

__global__ void deg_rsqrt(float* __restrict__ deg, int n) {
    int i = blockIdx.x * blockDim.x + threadIdx.x;
    if (i < n) deg[i] = rsqrtf(deg[i]);            // deg >= 1 always
}

// hs[i,c] = h[i,c] * dinv[i]; acc seeded with self-loop contribution hs[i,c].
__global__ void prescale_seed(const float* h, const float* __restrict__ dinv,
                              float* hs, float* acc, int total) {
    int idx = blockIdx.x * blockDim.x + threadIdx.x;
    if (idx >= total) return;
    float v = h[idx] * dinv[idx >> 6];
    hs[idx] = v;
    acc[idx] = v;
}

// acc[dst] += hs[src] : one 64-thread row group per edge, coalesced channels.
__global__ void edge_agg(const int* __restrict__ src, const int* __restrict__ dst,
                         const float* __restrict__ hs, float* acc, int E) {
    int e = blockIdx.x * blockDim.y + threadIdx.y;
    if (e >= E) return;
    int c = threadIdx.x;                           // 0..63
    int s = src[e];
    int d = dst[e];
    atomicAdd(&acc[(size_t)d * 64 + c], hs[(size_t)s * 64 + c]);
}

// out[i,c] = acc[i,c] * dinv[i] + bias[c]
__global__ void postscale_bias(const float* __restrict__ acc, const float* __restrict__ dinv,
                               const float* __restrict__ bias, float* __restrict__ out, int total) {
    int idx = blockIdx.x * blockDim.x + threadIdx.x;
    if (idx >= total) return;
    out[idx] = acc[idx] * dinv[idx >> 6] + bias[idx & 63];
}

// final 32 -> 1 projection
__global__ void final_dot(const float* __restrict__ m2, const float* __restrict__ lw3,
                          const float* __restrict__ lb3, float* __restrict__ out, int n) {
    int i = blockIdx.x * blockDim.x + threadIdx.x;
    if (i >= n) return;
    const float* r = m2 + (size_t)i * 32;
    float s = lb3[0];
#pragma unroll
    for (int c = 0; c < 32; ++c) s = fmaf(r[c], lw3[c], s);
    out[i] = s;
}

// ---------------------------------------------------------------------------
// B-fragment pre-pack: W[K_DIM x N_OUT] f32 row-major -> bf16 fragments in the
// exact 16x16x32 WMMA B layout, contiguous 16 elements per lane per tile:
//   frag[((kbi*NT + t)*32 + lane)*16 + e] = W[(kbi*32 + 8*(lane>=16) + (e&7) + 16*(e>=8))*N_OUT
//                                             + t*16 + lane%16]
// ---------------------------------------------------------------------------
__global__ void pack_b_frag(const float* __restrict__ W, __bf16* __restrict__ frag,
                            int K_DIM, int N_OUT) {
    int total = (K_DIM / 32) * (N_OUT / 16) * 512;
    int tid = blockIdx.x * blockDim.x + threadIdx.x;
    if (tid >= total) return;
    int e    = tid & 15;
    int lane = (tid >> 4) & 31;
    int tile = tid >> 9;
    int nt   = N_OUT / 16;
    int t    = tile % nt;
    int kbi  = tile / nt;
    int k = kbi * 32 + ((lane >> 4) << 3) + (e & 7) + ((e >> 3) << 4);
    int n = t * 16 + (lane & 15);
    frag[tid] = (__bf16)W[(size_t)k * N_OUT + n];
}

// ---------------------------------------------------------------------------
// WMMA bf16 GEMM:  Y[nrows x N_OUT] = X[nrows x K_DIM] @ W[K_DIM x N_OUT]
// One wave computes a (16*M_TILES)-row x N_OUT tile; B fragments pre-packed.
// ---------------------------------------------------------------------------

// A tile (16x32 bf16): lane L -> row L%16; element e -> K = e + 8*(e>=8) + 8*(L>=16)
// Loaded as 4x float4 (two 8-float runs) + pack to bf16.
__device__ inline v16bf load_a_tile(const float* __restrict__ X, int ldx,
                                    int row_base, int kbase, int lane, int nrows) {
    int m = row_base + (lane & 15);
    if (m >= nrows) m = nrows - 1;                 // clamp (keeps EXEC all-1s)
    const float4* xp = (const float4*)(X + (size_t)m * ldx + kbase + ((lane >> 4) << 3));
    float4 q0 = xp[0];                             // K offsets 0..3
    float4 q1 = xp[1];                             // 4..7
    float4 q2 = xp[4];                             // 16..19
    float4 q3 = xp[5];                             // 20..23
    v16bf a;
    a[0]  = (__bf16)q0.x; a[1]  = (__bf16)q0.y; a[2]  = (__bf16)q0.z; a[3]  = (__bf16)q0.w;
    a[4]  = (__bf16)q1.x; a[5]  = (__bf16)q1.y; a[6]  = (__bf16)q1.z; a[7]  = (__bf16)q1.w;
    a[8]  = (__bf16)q2.x; a[9]  = (__bf16)q2.y; a[10] = (__bf16)q2.z; a[11] = (__bf16)q2.w;
    a[12] = (__bf16)q3.x; a[13] = (__bf16)q3.y; a[14] = (__bf16)q3.z; a[15] = (__bf16)q3.w;
    return a;
}

template <int K_DIM, int N_OUT, int M_TILES, bool BIAS, bool RELU>
__global__ void gemm_wmma_bf16(const float* __restrict__ X, const __bf16* __restrict__ Bfrag,
                               const float* __restrict__ bias, float* __restrict__ Y, int nrows) {
    const int wave = threadIdx.x >> 5;
    const int lane = threadIdx.x & 31;
    const int unit = blockIdx.x * (blockDim.x >> 5) + wave;
    const int row0 = unit * (16 * M_TILES);
    if (row0 >= nrows) return;                     // wave-uniform exit

    constexpr int NT = N_OUT / 16;
    constexpr int KB = K_DIM / 32;
    v8f acc[M_TILES][NT] = {};

    const __bf16* bp = Bfrag + (size_t)lane * 16;

#pragma unroll
    for (int kbi = 0; kbi < KB; ++kbi) {
        v16bf a[M_TILES];
#pragma unroll
        for (int mt = 0; mt < M_TILES; ++mt)
            a[mt] = load_a_tile(X, K_DIM, row0 + mt * 16, kbi * 32, lane, nrows);
#pragma unroll
        for (int t = 0; t < NT; ++t) {
            v16bf b = *(const v16bf*)(bp + (size_t)(kbi * NT + t) * 512);
#pragma unroll
            for (int mt = 0; mt < M_TILES; ++mt)
                acc[mt][t] = __builtin_amdgcn_wmma_f32_16x16x32_bf16(
                    false, a[mt], false, b, (short)0, acc[mt][t], false, false);
        }
    }

    // C/D layout: element r of lane L -> (M = r + 8*(L>=16), N = L%16)
    const int nc = lane & 15;
    const int mo = (lane >> 4) << 3;
#pragma unroll
    for (int mt = 0; mt < M_TILES; ++mt) {
#pragma unroll
        for (int t = 0; t < NT; ++t) {
#pragma unroll
            for (int r = 0; r < 8; ++r) {
                int m = row0 + mt * 16 + mo + r;
                if (m < nrows) {
                    float v = acc[mt][t][r];
                    if constexpr (BIAS) v += bias[t * 16 + nc];
                    if constexpr (RELU) v = fmaxf(v, 0.0f);
                    Y[(size_t)m * N_OUT + t * 16 + nc] = v;
                }
            }
        }
    }
}

// ---------------------------------------------------------------------------
// Host-side launch
// ---------------------------------------------------------------------------

extern "C" void kernel_launch(void* const* d_in, const int* in_sizes, int n_in,
                              void* d_out, int out_size, void* d_ws, size_t ws_size,
                              hipStream_t stream) {
    const float* x   = (const float*)d_in[0];
    const int*   ei  = (const int*)d_in[1];
    const float* W1  = (const float*)d_in[2];
    const float* b1  = (const float*)d_in[3];
    const float* W2  = (const float*)d_in[4];
    const float* b2  = (const float*)d_in[5];
    const float* lw1 = (const float*)d_in[6];
    const float* lb1 = (const float*)d_in[7];
    const float* lw2 = (const float*)d_in[8];
    const float* lb2 = (const float*)d_in[9];
    const float* lw3 = (const float*)d_in[10];
    const float* lb3 = (const float*)d_in[11];
    float* out = (float*)d_out;

    const int n = in_sizes[0] / 128;               // 100000 (multiple of 32)
    const int E = in_sizes[1] / 2;                 // 3.2M
    const int* src = ei;
    const int* dst = ei + E;

    // workspace carve-out (256B aligned slices)
    char* wsp = (char*)d_ws;
    auto carve = [&](size_t bytes) { char* p = wsp; wsp += (bytes + 255) & ~(size_t)255; return p; };
    float*  dinv = (float*)carve((size_t)n * 4);
    float*  bufA = (float*)carve((size_t)n * 64 * 4);
    float*  bufB = (float*)carve((size_t)n * 64 * 4);
    __bf16* W1f  = (__bf16*)carve(128 * 64 * 2);   // fragment-packed
    __bf16* W2f  = (__bf16*)carve(64 * 64 * 2);
    __bf16* L1f  = (__bf16*)carve(64 * 64 * 2);
    __bf16* L2f  = (__bf16*)carve(64 * 32 * 2);

    const int tot64 = n * 64;
    const int tpb = 256;
    const int gemm_blocks = (n / 32 + 3) / 4;      // 4 waves/block, 32 rows/wave
    dim3 eb(64, 4);                                // 64 channels x 4 edges
    int  eg = (E + 3) / 4;

    // weights -> packed bf16 WMMA fragments
    pack_b_frag<<<(128 * 64 + tpb - 1) / tpb, tpb, 0, stream>>>(W1, W1f, 128, 64);
    pack_b_frag<<<(64 * 64 + tpb - 1) / tpb, tpb, 0, stream>>>(W2, W2f, 64, 64);
    pack_b_frag<<<(64 * 64 + tpb - 1) / tpb, tpb, 0, stream>>>(lw1, L1f, 64, 64);
    pack_b_frag<<<(64 * 32 + tpb - 1) / tpb, tpb, 0, stream>>>(lw2, L2f, 64, 32);

    // degrees -> dinv
    deg_init<<<(n + tpb - 1) / tpb, tpb, 0, stream>>>(dinv, n);
    deg_acc<<<(E + tpb - 1) / tpb, tpb, 0, stream>>>(dst, dinv, E);
    deg_rsqrt<<<(n + tpb - 1) / tpb, tpb, 0, stream>>>(dinv, n);

    // ---- GCN layer 1: h1 = x @ W1 ; aggregate ; *dinv + b1 ----
    gemm_wmma_bf16<128, 64, 2, false, false><<<gemm_blocks, 128, 0, stream>>>(x, W1f, nullptr, bufA, n);
    prescale_seed<<<(tot64 + tpb - 1) / tpb, tpb, 0, stream>>>(bufA, dinv, bufA, bufB, tot64);
    edge_agg<<<eg, eb, 0, stream>>>(src, dst, bufA, bufB, E);
    postscale_bias<<<(tot64 + tpb - 1) / tpb, tpb, 0, stream>>>(bufB, dinv, b1, bufA, tot64);

    // ---- GCN layer 2: h2 = h1 @ W2 ; aggregate ; *dinv + b2 ----
    gemm_wmma_bf16<64, 64, 2, false, false><<<gemm_blocks, 128, 0, stream>>>(bufA, W2f, nullptr, bufB, n);
    prescale_seed<<<(tot64 + tpb - 1) / tpb, tpb, 0, stream>>>(bufB, dinv, bufB, bufA, tot64);
    edge_agg<<<eg, eb, 0, stream>>>(src, dst, bufB, bufA, E);
    postscale_bias<<<(tot64 + tpb - 1) / tpb, tpb, 0, stream>>>(bufA, dinv, b2, bufB, tot64);

    // ---- MLP head ----
    gemm_wmma_bf16<64, 64, 2, true, true><<<gemm_blocks, 128, 0, stream>>>(bufB, L1f, lb1, bufA, n);
    gemm_wmma_bf16<64, 32, 2, true, true><<<gemm_blocks, 128, 0, stream>>>(bufA, L2f, lb2, bufB, n);
    final_dot<<<(n + tpb - 1) / tpb, tpb, 0, stream>>>(bufB, lw3, lb3, out, n);
}
